// CasualMultiHeadAttention_28303834480703
// MI455X (gfx1250) — matrix operand order repaired and checked by reading, not compiled, not run on hardware
//
#include <hip/hip_runtime.h>

// ---------------------------------------------------------------------------
// Causal MHA for MI455X (gfx1250): f16 WMMA (16x16x32), f32 accumulate.
// B=2, T=2048, C=1024, H=16, Dh=64.
// Round 2: LDS-transposed b128 store paths, M=32 flash tiles (2x arithmetic
// intensity on K/V), interleaved independent WMMAs to kill hazard NOPs.
// ---------------------------------------------------------------------------

typedef __attribute__((ext_vector_type(16))) _Float16 v16h;
typedef __attribute__((ext_vector_type(8)))  _Float16 v8h;
typedef __attribute__((ext_vector_type(8)))  float    v8f;
typedef __attribute__((ext_vector_type(4)))  float    f32x4;

#define WMMA_F16(A, B, C) \
  __builtin_amdgcn_wmma_f32_16x16x32_f16(false, (A), false, (B), (short)0, (C), false, false)

constexpr int T_ = 2048;
constexpr int C_ = 1024;
constexpr int D_ = 64;   // head dim

static __device__ inline v8f zero8() {
  v8f z = {0.f, 0.f, 0.f, 0.f, 0.f, 0.f, 0.f, 0.f};
  return z;
}

static __device__ inline v16h cat8(v8h lo, v8h hh) {
  v16h r;
#pragma unroll
  for (int j = 0; j < 8; ++j) { r[j] = lo[j]; r[8 + j] = hh[j]; }
  return r;
}

static __device__ inline v16h cvt16(f32x4 a, f32x4 b, f32x4 c, f32x4 d) {
  v16h r;
#pragma unroll
  for (int j = 0; j < 4; ++j) {
    r[j]      = (_Float16)a[j];
    r[4 + j]  = (_Float16)b[j];
    r[8 + j]  = (_Float16)c[j];
    r[12 + j] = (_Float16)d[j];
  }
  return r;
}

// ---------------------------------------------------------------------------
// Kernel 0: Wo (C x C f32, row-major) -> f16 row-major (B-operand friendly).
// ---------------------------------------------------------------------------
__global__ __launch_bounds__(256) void k_wo_cvt(const float* __restrict__ wo,
                                                _Float16* __restrict__ wo16) {
  const int i = (blockIdx.x * 256 + threadIdx.x) * 4;
  f32x4 v = *(const f32x4*)(wo + i);
#pragma unroll
  for (int j = 0; j < 4; ++j) wo16[i + j] = (_Float16)v[j];
}

// ---------------------------------------------------------------------------
// Kernel 1: fused QKV projection. One wave = one (b,h, 16-row t-tile).
// q,k stored (BH,T,Dh) f16 via LDS-transpose + b128 stores.
// v stored transposed (BH,Dh,T) f16 via packed b128 stores.
// ---------------------------------------------------------------------------
__global__ __launch_bounds__(256) void k_qkv(const float* __restrict__ x,
                                             const float* __restrict__ Wq,
                                             const float* __restrict__ Wk,
                                             const float* __restrict__ Wv,
                                             _Float16* __restrict__ qh,
                                             _Float16* __restrict__ kh,
                                             _Float16* __restrict__ vt) {
  __shared__ _Float16 tbuf[8][16 * 64];     // per-wave transpose staging (2KB)
  const int wslot = threadIdx.x >> 5;
  const int wid   = blockIdx.x * 8 + wslot;
  const int lane  = threadIdx.x & 31;
  const int tt = wid & 127;                 // T/16 = 128 tiles
  const int bh = wid >> 7;                  // 0..31
  const int t0 = tt << 4;
  const int b = bh >> 4, h = bh & 15;
  const int r = lane & 15, hi = lane >> 4, kb = hi << 4;

  // A operand: x tile 16x64 -> f16, two K=32 chunks
  v16h a[2];
#pragma unroll
  for (int c = 0; c < 2; ++c) {
    const float* px = x + ((size_t)(b * T_ + t0 + r)) * C_ + h * D_ + c * 32 + kb;
    a[c] = cvt16(*(const f32x4*)(px + 0),  *(const f32x4*)(px + 4),
                 *(const f32x4*)(px + 8), *(const f32x4*)(px + 12));
  }

  v8f accq[4], acck[4], accv[4];
#pragma unroll
  for (int e4 = 0; e4 < 4; ++e4) { accq[e4] = zero8(); acck[e4] = zero8(); accv[e4] = zero8(); }

#pragma unroll
  for (int e4 = 0; e4 < 4; ++e4) {
    const int e0 = e4 << 4;
#pragma unroll
    for (int c = 0; c < 2; ++c) {
      const size_t wofs = (size_t)(e0 + r) * D_ + c * 32 + hi * 16;
      v16h bq = cvt16(*(const f32x4*)(Wq + wofs + 0), *(const f32x4*)(Wq + wofs + 4),
                      *(const f32x4*)(Wq + wofs + 8), *(const f32x4*)(Wq + wofs + 12));
      v16h bk = cvt16(*(const f32x4*)(Wk + wofs + 0), *(const f32x4*)(Wk + wofs + 4),
                      *(const f32x4*)(Wk + wofs + 8), *(const f32x4*)(Wk + wofs + 12));
      v16h bv = cvt16(*(const f32x4*)(Wv + wofs + 0), *(const f32x4*)(Wv + wofs + 4),
                      *(const f32x4*)(Wv + wofs + 8), *(const f32x4*)(Wv + wofs + 12));
      // three independent WMMAs back-to-back (no RAW hazard NOPs)
      accq[e4] = WMMA_F16(a[c], bq, accq[e4]);
      acck[e4] = WMMA_F16(a[c], bk, acck[e4]);
      accv[e4] = WMMA_F16(a[c], bv, accv[e4]);
    }
  }

  // ---- v: transposed layout is contiguous per lane -> one b128 per chunk ----
#pragma unroll
  for (int e4 = 0; e4 < 4; ++e4) {
    v8h pk8;
#pragma unroll
    for (int i = 0; i < 8; ++i) pk8[i] = (_Float16)accv[e4][i];
    *(v8h*)(vt + ((size_t)bh * D_ + (e4 << 4) + r) * T_ + t0 + (hi << 3)) = pk8;
  }

  // ---- q, k: LDS transpose -> contiguous 32B row stores ----
  _Float16* tb = tbuf[wslot];
#pragma unroll
  for (int e4 = 0; e4 < 4; ++e4)
#pragma unroll
    for (int i = 0; i < 8; ++i)
      tb[(((hi << 3) + i) << 6) + (e4 << 4) + r] = (_Float16)accq[e4][i];
  *(v16h*)(qh + ((size_t)bh * T_ + t0 + r) * D_ + (hi << 5)) =
      *(const v16h*)(tb + (r << 6) + (hi << 5));

#pragma unroll
  for (int e4 = 0; e4 < 4; ++e4)
#pragma unroll
    for (int i = 0; i < 8; ++i)
      tb[(((hi << 3) + i) << 6) + (e4 << 4) + r] = (_Float16)acck[e4][i];
  *(v16h*)(kh + ((size_t)bh * T_ + t0 + r) * D_ + (hi << 5)) =
      *(const v16h*)(tb + (r << 6) + (hi << 5));
}

// ---------------------------------------------------------------------------
// Kernel 2: causal flash attention, M=32 query rows per wave (two 16-row
// subtiles sharing every K/V B-operand load), 32-wide KV blocks, online
// softmax. Consecutive waves of a block work on consecutive q-tiles of the
// same (b,h) -> shared KV working set in WGP$/L2.
// ---------------------------------------------------------------------------
__global__ __launch_bounds__(256) void k_flash(const _Float16* __restrict__ qh,
                                               const _Float16* __restrict__ kh,
                                               const _Float16* __restrict__ vt,
                                               _Float16* __restrict__ att) {
  __shared__ _Float16 sbuf[8][2048];        // per-wave staging (4KB)
  const int wslot = threadIdx.x >> 5;
  const int wid   = blockIdx.x * 8 + wslot;
  const int lane  = threadIdx.x & 31;
  const int tt = wid & 63;                  // T/32 = 64 q-tiles per (b,h)
  const int bh = wid >> 6;
  const int qt0 = tt << 5;
  const int r = lane & 15, hi = lane >> 4, kb = hi << 4;

  // Q A-operands: [qtile][k-chunk]
  v16h aq[2][2];
#pragma unroll
  for (int t = 0; t < 2; ++t)
#pragma unroll
    for (int c = 0; c < 2; ++c) {
      const _Float16* pq = qh + ((size_t)bh * T_ + qt0 + t * 16 + r) * D_ + c * 32;
      aq[t][c] = cat8(*(const v8h*)(pq + kb), *(const v8h*)(pq + kb + 8));
    }

  float mrow[2][8], lrow[2][8];
  v8f o[2][4];
#pragma unroll
  for (int t = 0; t < 2; ++t) {
#pragma unroll
    for (int i = 0; i < 8; ++i) { mrow[t][i] = -3.0e38f; lrow[t][i] = 0.f; }
#pragma unroll
    for (int c4 = 0; c4 < 4; ++c4) o[t][c4] = zero8();
  }

  _Float16* sw = sbuf[wslot];

  for (int kv0 = 0; kv0 <= qt0 + 31; kv0 += 32) {
    // ---- S = Q K^T : 8 WMMAs, B operands loaded once, reused for both q-tiles
    const _Float16* pk0 = kh + ((size_t)bh * T_ + kv0 + r) * D_;
    const _Float16* pk1 = kh + ((size_t)bh * T_ + kv0 + 16 + r) * D_;
    v8f s[2][2];
#pragma unroll
    for (int c = 0; c < 2; ++c) {
      v16h bk0 = *(const v16h*)(pk0 + c * 32 + hi * 16);
      v16h bk1 = *(const v16h*)(pk1 + c * 32 + hi * 16);
      if (c == 0) {
        s[0][0] = WMMA_F16(aq[0][0], bk0, zero8());
        s[0][1] = WMMA_F16(aq[0][0], bk1, zero8());
        s[1][0] = WMMA_F16(aq[1][0], bk0, zero8());
        s[1][1] = WMMA_F16(aq[1][0], bk1, zero8());
      } else {
        s[0][0] = WMMA_F16(aq[0][1], bk0, s[0][0]);
        s[0][1] = WMMA_F16(aq[0][1], bk1, s[0][1]);
        s[1][0] = WMMA_F16(aq[1][1], bk0, s[1][0]);
        s[1][1] = WMMA_F16(aq[1][1], bk1, s[1][1]);
      }
    }

    v16h ap[2];
#pragma unroll
    for (int t = 0; t < 2; ++t) {
      // ---- causal mask + 1/sqrt(Dh) scale ----
#pragma unroll
      for (int st = 0; st < 2; ++st) {
        const int kv = kv0 + st * 16 + r;
#pragma unroll
        for (int i = 0; i < 8; ++i) {
          const int row = qt0 + t * 16 + (hi << 3) + i;
          s[t][st][i] = (kv <= row) ? s[t][st][i] * 0.125f : -3.0e38f;
        }
      }
      // ---- row max / rescale ----
      float mnew[8], alpha[8];
#pragma unroll
      for (int i = 0; i < 8; ++i) {
        float v = fmaxf(s[t][0][i], s[t][1][i]);
        v = fmaxf(v, __shfl_xor(v, 1, 32));
        v = fmaxf(v, __shfl_xor(v, 2, 32));
        v = fmaxf(v, __shfl_xor(v, 4, 32));
        v = fmaxf(v, __shfl_xor(v, 8, 32));   // reduce within 16-lane half
        mnew[i]  = fmaxf(mrow[t][i], v);
        alpha[i] = __expf(mrow[t][i] - mnew[i]);
        mrow[t][i] = mnew[i];
      }
      // ---- P = exp(S-m), stage to LDS (C-layout -> row-major 16x32) ----
      _Float16* pP = sw + t * 512;
#pragma unroll
      for (int st = 0; st < 2; ++st)
#pragma unroll
        for (int i = 0; i < 8; ++i) {
          float p = __expf(s[t][st][i] - mnew[i]);
          s[t][st][i] = p;
          pP[(((hi << 3) + i) << 5) + st * 16 + r] = (_Float16)p;
        }
      // ---- row sums ----
#pragma unroll
      for (int i = 0; i < 8; ++i) {
        float rs = s[t][0][i] + s[t][1][i];
        rs += __shfl_xor(rs, 1, 32);
        rs += __shfl_xor(rs, 2, 32);
        rs += __shfl_xor(rs, 4, 32);
        rs += __shfl_xor(rs, 8, 32);
        lrow[t][i] = lrow[t][i] * alpha[i] + rs;
      }
      // ---- rescale O ----
#pragma unroll
      for (int c4 = 0; c4 < 4; ++c4)
#pragma unroll
        for (int i = 0; i < 8; ++i) o[t][c4][i] *= alpha[i];
      // ---- P back in A-operand layout ----
      ap[t] = cat8(*(const v8h*)(pP + (r << 5) + kb),
                   *(const v8h*)(pP + (r << 5) + kb + 8));
    }

    // ---- O += P V : 4 V loads shared by both q-tiles, 8 WMMAs interleaved ----
#pragma unroll
    for (int c4 = 0; c4 < 4; ++c4) {
      v16h bv = *(const v16h*)(vt + ((size_t)bh * D_ + c4 * 16 + r) * T_ + kv0 + hi * 16);
      o[0][c4] = WMMA_F16(ap[0], bv, o[0][c4]);
      o[1][c4] = WMMA_F16(ap[1], bv, o[1][c4]);
    }
  }

  // ---- epilogue: 1/l, LDS transpose, contiguous 32B stores ----
  const int b = bh >> 4, h = bh & 15;
#pragma unroll
  for (int t = 0; t < 2; ++t) {
    _Float16* pO = sw + t * 1024;           // 16x64 halves, row stride 64
#pragma unroll
    for (int i = 0; i < 8; ++i) {
      const float inv = 1.0f / lrow[t][i];
#pragma unroll
      for (int c4 = 0; c4 < 4; ++c4)
        pO[(((hi << 3) + i) << 6) + c4 * 16 + r] = (_Float16)(o[t][c4][i] * inv);
    }
  }
#pragma unroll
  for (int t = 0; t < 2; ++t) {
    const _Float16* pO = sw + t * 1024;
    *(v16h*)(att + ((size_t)(b * T_ + qt0 + t * 16 + r)) * C_ + h * D_ + (hi << 5)) =
        *(const v16h*)(pO + (r << 6) + (hi << 5));
  }
}

// ---------------------------------------------------------------------------
// Kernel 3: Y = att (4096 x 1024) @ Wo^T, f32 out. One wave = 16x64 tile,
// LDS f32 transpose -> contiguous b128 stores.
// ---------------------------------------------------------------------------
__global__ __launch_bounds__(256) void k_proj(const _Float16* __restrict__ att,
                                              const _Float16* __restrict__ wo16,
                                              float* __restrict__ out) {
  __shared__ float fbuf[8][16 * 64];        // per-wave 4KB
  const int wslot = threadIdx.x >> 5;
  const int wid   = blockIdx.x * 8 + wslot;
  const int lane  = threadIdx.x & 31;
  const int rt = wid >> 4;                  // 256 row tiles
  const int e0 = (wid & 15) << 6;
  const int r = lane & 15, hi = lane >> 4, kb = hi << 4;

  v8f o[4];
#pragma unroll
  for (int c4 = 0; c4 < 4; ++c4) o[c4] = zero8();

  for (int kc = 0; kc < C_; kc += 32) {
    const _Float16* pa = att + (size_t)(rt * 16 + r) * C_ + kc;
    v16h ap = cat8(*(const v8h*)(pa + kb), *(const v8h*)(pa + kb + 8));
#pragma unroll
    for (int s4 = 0; s4 < 4; ++s4) {
      const _Float16* pw = wo16 + (size_t)(e0 + s4 * 16 + r) * C_ + kc + hi * 16;
      o[s4] = WMMA_F16(ap, *(const v16h*)(pw), o[s4]);   // 4 independent accums
    }
  }

  float* fb = fbuf[wslot];
#pragma unroll
  for (int s4 = 0; s4 < 4; ++s4)
#pragma unroll
    for (int i = 0; i < 8; ++i)
      fb[(((hi << 3) + i) << 6) + s4 * 16 + r] = o[s4][i];
#pragma unroll
  for (int u = 0; u < 8; ++u) {
    f32x4 v = *(const f32x4*)(fb + (r << 6) + (hi << 5) + u * 4);
    *(f32x4*)(out + (size_t)(rt * 16 + r) * C_ + e0 + (hi << 5) + u * 4) = v;
  }
}

// ---------------------------------------------------------------------------
// Host launcher.  Workspace map (bytes):
//   [0,8M)    q  f16 (BH,T,Dh)    [8M,16M)  k  f16 (BH,T,Dh)
//   [16M,24M) vT f16 (BH,Dh,T)    [24M,32M) att f16 (B,T,C)
//   [32M,34M) Wo f16 (C,C)
// ---------------------------------------------------------------------------
extern "C" void kernel_launch(void* const* d_in, const int* in_sizes, int n_in,
                              void* d_out, int out_size, void* d_ws, size_t ws_size,
                              hipStream_t stream) {
  const float* x  = (const float*)d_in[0];
  const float* Wq = (const float*)d_in[1];
  const float* Wk = (const float*)d_in[2];
  const float* Wv = (const float*)d_in[3];
  const float* Wo = (const float*)d_in[4];
  float* out = (float*)d_out;

  char* ws = (char*)d_ws;
  _Float16* qh   = (_Float16*)(ws);
  _Float16* kh   = (_Float16*)(ws + (size_t)8  * 1024 * 1024);
  _Float16* vt   = (_Float16*)(ws + (size_t)16 * 1024 * 1024);
  _Float16* att  = (_Float16*)(ws + (size_t)24 * 1024 * 1024);
  _Float16* wo16 = (_Float16*)(ws + (size_t)32 * 1024 * 1024);

  k_wo_cvt<<<1024, 256, 0, stream>>>(Wo, wo16);            // C*C/(256*4)
  k_qkv  <<<512, 256, 0, stream>>>(x, Wq, Wk, Wv, qh, kh, vt);  // 4096 waves
  k_flash<<<256, 256, 0, stream>>>(qh, kh, vt, att);       // 2048 waves (M=32)
  k_proj <<<512, 256, 0, stream>>>(att, wo16, out);        // 4096 waves
}
